// CrossEntropy_28475633172840
// MI455X (gfx1250) — matrix operand order repaired and checked by reading, not compiled, not run on hardware
//
#include <hip/hip_runtime.h>

// CDNA5 / gfx1250 graph-embedding BCE-with-logits loss.
//
// Roofline: ~0.82 GFLOP vs ~1.6GB (f32) of L2-resident gather traffic ->
// strongly bandwidth bound. Strategy:
//   1) one-shot convert node_features f32 -> f16 into workspace (L2-resident,
//      halves gathered bytes and matches WMMA input type),
//   2) one wave32 per node: A = 16 gathered neighbor rows (f16),
//      B = node feature chunk replicated across all 16 columns,
//      4 chained v_wmma_f32_16x16x32_f16 over K=128 -> D[m,*] = <nb_m, x_n>,
//   3) softplus(+-D) summed over the whole fragment = 16 * per-node loss sum
//      (columns are identical replicas, nonlinearity commutes with that),
//   4) deterministic two-stage reduction (no float atomics).
// Falls back to f32-gather path if ws_size can't hold the f16 matrix.

typedef __attribute__((ext_vector_type(16))) _Float16 v16h;
typedef __attribute__((ext_vector_type(8)))  _Float16 v8h;
typedef __attribute__((ext_vector_type(4)))  _Float16 v4h;
typedef __attribute__((ext_vector_type(8)))  float    v8f;

#define D_FEAT 128
#define K_NEI  16          // neighbors per tile (pos tile / neg tile)
#define WAVES_PER_BLOCK 8  // 256 threads = 8 wave32 = 8 nodes per block

__device__ __forceinline__ float softplus_stable(float x) {
  // log(1 + exp(x)), numerically stable
  return log1pf(__expf(-fabsf(x))) + fmaxf(x, 0.0f);
}

// ---------------------------------------------------------------------------
// One-shot f32 -> f16 conversion of the feature matrix (streams through HBM/L2
// once; after this the 25.6MB f16 matrix stays L2-resident).
// ---------------------------------------------------------------------------
__global__ __launch_bounds__(256)
void cvt_f32_to_f16_kernel(const float* __restrict__ in,
                           _Float16* __restrict__ out, int total4) {
  int i = blockIdx.x * blockDim.x + threadIdx.x;
  if (i < total4) {
    float4 f = reinterpret_cast<const float4*>(in)[i];
    v4h h;
    h[0] = (_Float16)f.x; h[1] = (_Float16)f.y;
    h[2] = (_Float16)f.z; h[3] = (_Float16)f.w;
    reinterpret_cast<v4h*>(out)[i] = h;
  }
}

// ---------------------------------------------------------------------------
// Fast path: gather f16 rows directly into WMMA fragments.
// A fragment (16-bit A 16x32, contiguous k = 16h+s, not the ISA table order): lane l owns row (l&15);
//   lanes 0-15 : a[0..15]=K(kb+0..15)
//   lanes 16-31: a[0..15]=K(kb+16..31)
// B fragment (16-bit B 32x16, same contiguous k order as A): lane l owns col (l&15);
//   lanes 0-15 hold K(kb+0..15), lanes 16-31 hold K(kb+16..31).
// ---------------------------------------------------------------------------
__device__ __forceinline__ v8f dot_tile_h(const _Float16* __restrict__ feat,
                                          const _Float16* __restrict__ xn,
                                          const int*      __restrict__ idx16,
                                          int lane) {
  const int row  = lane & 15;
  const int half = lane >> 4;
  const _Float16* __restrict__ arow =
      feat + (long long)idx16[row] * (long long)D_FEAT;

  v8f c = {};
#pragma unroll
  for (int ck = 0; ck < 4; ++ck) {
    const int kb = ck * 32;
    v8h alo = *reinterpret_cast<const v8h*>(arow + kb + half * 16);       // 16B
    v8h ahi = *reinterpret_cast<const v8h*>(arow + kb + half * 16 + 8);   // 16B
    v16h a  = __builtin_shufflevector(alo, ahi,
                0, 1, 2, 3, 4, 5, 6, 7, 8, 9, 10, 11, 12, 13, 14, 15);
    v16h b  = *reinterpret_cast<const v16h*>(xn + kb + half * 16);        // 32B bcast
    c = __builtin_amdgcn_wmma_f32_16x16x32_f16(
        false, a, false, b, (short)0, c, false, false);
  }
  return c;
}

__global__ __launch_bounds__(256)
void gsl_bce_wmma_h_kernel(const _Float16* __restrict__ feat,
                           const int*      __restrict__ pos_idx,
                           const int*      __restrict__ neg_idx,
                           float*          __restrict__ block_partials,
                           int N) {
  __shared__ float sdata[WAVES_PER_BLOCK];
  const int lane = threadIdx.x & 31;
  const int wave = threadIdx.x >> 5;
  const int node = blockIdx.x * WAVES_PER_BLOCK + wave;

  float partial = 0.0f;
  if (node < N) {  // wave-uniform: EXEC stays all-ones inside (WMMA requirement)
    const _Float16* xn = feat + (long long)node * D_FEAT;
    __builtin_prefetch(xn, 0, 3);  // global_prefetch_b8

    v8f dpos = dot_tile_h(feat, xn, pos_idx + node * K_NEI, lane);
    v8f dneg = dot_tile_h(feat, xn, neg_idx + node * K_NEI, lane);

    float s = 0.0f;
#pragma unroll
    for (int i = 0; i < 8; ++i) {
      s += softplus_stable(-dpos[i]);  // label 1: softplus(-s)
      s += softplus_stable( dneg[i]);  // label 0: softplus(+s)
    }
#pragma unroll
    for (int off = 16; off > 0; off >>= 1)
      s += __shfl_down(s, off, 32);
    partial = s;  // lane 0: 16 * sum_k loss_k for this node
  }

  if (lane == 0) sdata[wave] = partial;
  __syncthreads();
  if (threadIdx.x == 0) {
    float t = 0.0f;
#pragma unroll
    for (int w = 0; w < WAVES_PER_BLOCK; ++w) t += sdata[w];
    block_partials[blockIdx.x] = t;
  }
}

// ---------------------------------------------------------------------------
// Fallback path (ws too small for f16 matrix): gather f32, convert in-loop.
// ---------------------------------------------------------------------------
__device__ __forceinline__ v8f dot_tile_f(const float* __restrict__ feat,
                                          const float* __restrict__ xn,
                                          const int*   __restrict__ idx16,
                                          int lane) {
  const int row  = lane & 15;
  const int half = lane >> 4;
  const float* __restrict__ arow =
      feat + (long long)idx16[row] * (long long)D_FEAT;

  v8f c = {};
#pragma unroll
  for (int ck = 0; ck < 4; ++ck) {
    const int kb = ck * 32;
    const float4* pa0 = reinterpret_cast<const float4*>(arow + kb + half * 16);
    const float4* pa1 = reinterpret_cast<const float4*>(arow + kb + half * 16 + 8);
    float4 f0 = pa0[0], f1 = pa0[1], f2 = pa1[0], f3 = pa1[1];
    const float4* pb = reinterpret_cast<const float4*>(xn + kb + half * 16);
    float4 g0 = pb[0], g1 = pb[1], g2 = pb[2], g3 = pb[3];

    v16h a, b;
    a[0]=(_Float16)f0.x; a[1]=(_Float16)f0.y; a[2]=(_Float16)f0.z; a[3]=(_Float16)f0.w;
    a[4]=(_Float16)f1.x; a[5]=(_Float16)f1.y; a[6]=(_Float16)f1.z; a[7]=(_Float16)f1.w;
    a[8]=(_Float16)f2.x; a[9]=(_Float16)f2.y; a[10]=(_Float16)f2.z; a[11]=(_Float16)f2.w;
    a[12]=(_Float16)f3.x; a[13]=(_Float16)f3.y; a[14]=(_Float16)f3.z; a[15]=(_Float16)f3.w;
    b[0]=(_Float16)g0.x; b[1]=(_Float16)g0.y; b[2]=(_Float16)g0.z; b[3]=(_Float16)g0.w;
    b[4]=(_Float16)g1.x; b[5]=(_Float16)g1.y; b[6]=(_Float16)g1.z; b[7]=(_Float16)g1.w;
    b[8]=(_Float16)g2.x; b[9]=(_Float16)g2.y; b[10]=(_Float16)g2.z; b[11]=(_Float16)g2.w;
    b[12]=(_Float16)g3.x; b[13]=(_Float16)g3.y; b[14]=(_Float16)g3.z; b[15]=(_Float16)g3.w;

    c = __builtin_amdgcn_wmma_f32_16x16x32_f16(
        false, a, false, b, (short)0, c, false, false);
  }
  return c;
}

__global__ __launch_bounds__(256)
void gsl_bce_wmma_f_kernel(const float* __restrict__ feat,
                           const int*   __restrict__ pos_idx,
                           const int*   __restrict__ neg_idx,
                           float*       __restrict__ block_partials,
                           int N) {
  __shared__ float sdata[WAVES_PER_BLOCK];
  const int lane = threadIdx.x & 31;
  const int wave = threadIdx.x >> 5;
  const int node = blockIdx.x * WAVES_PER_BLOCK + wave;

  float partial = 0.0f;
  if (node < N) {
    const float* xn = feat + (long long)node * D_FEAT;
    v8f dpos = dot_tile_f(feat, xn, pos_idx + node * K_NEI, lane);
    v8f dneg = dot_tile_f(feat, xn, neg_idx + node * K_NEI, lane);
    float s = 0.0f;
#pragma unroll
    for (int i = 0; i < 8; ++i) {
      s += softplus_stable(-dpos[i]);
      s += softplus_stable( dneg[i]);
    }
#pragma unroll
    for (int off = 16; off > 0; off >>= 1)
      s += __shfl_down(s, off, 32);
    partial = s;
  }

  if (lane == 0) sdata[wave] = partial;
  __syncthreads();
  if (threadIdx.x == 0) {
    float t = 0.0f;
#pragma unroll
    for (int w = 0; w < WAVES_PER_BLOCK; ++w) t += sdata[w];
    block_partials[blockIdx.x] = t;
  }
}

// ---------------------------------------------------------------------------
// Deterministic final reduction.
// ---------------------------------------------------------------------------
__global__ __launch_bounds__(256)
void gsl_reduce_kernel(const float* __restrict__ partials, int nparts,
                       float* __restrict__ out, float scale) {
  __shared__ float sdata[256];
  float s = 0.0f;
  for (int i = threadIdx.x; i < nparts; i += 256) s += partials[i];  // fixed order
  sdata[threadIdx.x] = s;
  __syncthreads();
#pragma unroll
  for (int stride = 128; stride > 0; stride >>= 1) {
    if ((int)threadIdx.x < stride) sdata[threadIdx.x] += sdata[threadIdx.x + stride];
    __syncthreads();
  }
  if (threadIdx.x == 0) out[0] = sdata[0] * scale;
}

extern "C" void kernel_launch(void* const* d_in, const int* in_sizes, int n_in,
                              void* d_out, int out_size, void* d_ws, size_t ws_size,
                              hipStream_t stream) {
  const float* feat = (const float*)d_in[0];
  const int*   pos  = (const int*)d_in[1];   // JAX default x64-off => int32
  const int*   neg  = (const int*)d_in[2];

  const int N = in_sizes[0] / D_FEAT;
  const int nblocks = (N + WAVES_PER_BLOCK - 1) / WAVES_PER_BLOCK;
  // Each node contributes 16 * sum_{k=0..31} loss_k; divide by the fragment
  // redundancy (16), k_total (32), and N for the mean.
  const float scale = 1.0f / (16.0f * 32.0f * (float)N);

  // ws layout (fast path): [ f16 features: N*D*2 B (256B-aligned pad) | partials ]
  const size_t featBytes = (size_t)N * D_FEAT * sizeof(_Float16);
  const size_t featPad   = (featBytes + 255) & ~(size_t)255;
  const size_t needFast  = featPad + (size_t)nblocks * sizeof(float);

  if (ws_size >= needFast) {
    _Float16* feat16  = (_Float16*)d_ws;
    float* partials   = (float*)((char*)d_ws + featPad);
    const int total4  = (N * D_FEAT) / 4;
    cvt_f32_to_f16_kernel<<<(total4 + 255) / 256, 256, 0, stream>>>(
        feat, feat16, total4);
    gsl_bce_wmma_h_kernel<<<nblocks, 256, 0, stream>>>(
        feat16, pos, neg, partials, N);
    gsl_reduce_kernel<<<1, 256, 0, stream>>>(partials, nblocks,
                                             (float*)d_out, scale);
  } else {
    float* partials = (float*)d_ws;  // nblocks floats
    gsl_bce_wmma_f_kernel<<<nblocks, 256, 0, stream>>>(
        feat, pos, neg, partials, N);
    gsl_reduce_kernel<<<1, 256, 0, stream>>>(partials, nblocks,
                                             (float*)d_out, scale);
  }
}
